// MultiHeadAttention_26792005992736
// MI455X (gfx1250) — matrix operand order repaired and checked by reading, not compiled, not run on hardware
//
#include <hip/hip_runtime.h>

typedef unsigned short u16;
typedef __attribute__((ext_vector_type(16))) __bf16 v16bf;
typedef __attribute__((ext_vector_type(8)))  float  v8f;
typedef int vec4i __attribute__((vector_size(16)));

#define S_LEN   2048
#define D_MODEL 1024
#define NHEADS  16
#define HDIM    64
#define BATCH   2
#define M_TOT   (BATCH * S_LEN)   /* 4096 */

#define AS1 __attribute__((address_space(1)))
#define AS3 __attribute__((address_space(3)))

#if defined(__has_builtin)
#if __has_builtin(__builtin_amdgcn_global_load_async_to_lds_b128)
#define HAVE_ASYNC_LDS 1
#endif
#endif
#ifndef HAVE_ASYNC_LDS
#define HAVE_ASYNC_LDS 0
#endif

__device__ __forceinline__ u16 f32_to_bf16(float f) {
  unsigned u = __float_as_uint(f);
  u += 0x7FFFu + ((u >> 16) & 1u);   // round-to-nearest-even
  return (u16)(u >> 16);
}

union FragAB { v16bf v; uint4 u[2]; };
union Acc8   { v8f  v; float f[8]; };

// 16B global -> LDS copy: async DMA (ASYNCcnt) when available, else via VGPRs
__device__ __forceinline__ void cp_g2l_b128(const u16* g, u16* l) {
#if HAVE_ASYNC_LDS
  __builtin_amdgcn_global_load_async_to_lds_b128(
      (AS1 vec4i*)(vec4i*)(u16*)g, (AS3 vec4i*)(vec4i*)l, 0, 0);
#else
  *(uint4*)l = *(const uint4*)g;
#endif
}
__device__ __forceinline__ void async_wait() {
#if HAVE_ASYNC_LDS
#if __has_builtin(__builtin_amdgcn_s_wait_asynccnt)
  __builtin_amdgcn_s_wait_asynccnt(0);
#else
  asm volatile("s_wait_asynccnt 0" ::: "memory");
#endif
#endif
}

// ---------------------------------------------------------------- convert
__global__ void cvt_f32_bf16(const float* __restrict__ in, u16* __restrict__ out, int n) {
  int i = blockIdx.x * 256 + threadIdx.x;
  if (i < n) out[i] = f32_to_bf16(in[i]);
}

// in: [Kd, N] row-major (fp32)  ->  out: [N, Kd] row-major (bf16)
__global__ void transpose_to_bf16(const float* __restrict__ in, u16* __restrict__ out,
                                  int N, int Kd) {
  int i = blockIdx.x * 256 + threadIdx.x;
  if (i < N * Kd) {
    int nrow = i / Kd, k = i % Kd;
    out[i] = f32_to_bf16(in[k * N + nrow]);
  }
}

// ---------------------------------------------------------------- GEMM
// C[M,Ntot] = A[M,1024] @ Bt[Ntot,1024]^T (+bias).
// 256 threads = 8 waves; block tile 128x128; wave grid 4(M) x 2(N);
// wave tile 32x64 -> 8 f32 accumulators; K staged 64 per barrier.
// mode 0: scatter into Q[B,H,S,hd], K[B,H,S,hd], Vt[B,H,hd,S] (bf16, +b_exp)
// mode 1: write fp32 out[M,1024] (+b_comp)
__global__ void gemm_bf16(const u16* __restrict__ A, const u16* __restrict__ Bt,
                          const float* __restrict__ bias, int mode,
                          u16* __restrict__ Qo, u16* __restrict__ Ko,
                          u16* __restrict__ Vto, float* __restrict__ out) {
  const int Dk = D_MODEL;
  __shared__ u16 At[128 * 72];   // 128 rows, 64 K + 8 pad
  __shared__ u16 Bs[128 * 72];

  const int tid = threadIdx.x;
  const int wave = tid >> 5, lane = tid & 31;
  const int lane16 = lane & 15, grp = lane >> 4, grp8 = grp * 8;
  const int wm = (wave >> 1) * 32, wn = (wave & 1) * 64;
  const int n0 = blockIdx.x * 128, m0 = blockIdx.y * 128;

  Acc8 acc[2][4];
#pragma unroll
  for (int ms = 0; ms < 2; ++ms)
#pragma unroll
    for (int ns = 0; ns < 4; ++ns)
#pragma unroll
      for (int r = 0; r < 8; ++r) acc[ms][ns].f[r] = 0.f;

  const int srow = tid >> 1, scol0 = (tid & 1) * 32;   // 2 thr/row, 64B each
  for (int kc = 0; kc < Dk; kc += 64) {
    const u16* ga = A  + (size_t)(m0 + srow) * Dk + kc + scol0;
    const u16* gb = Bt + (size_t)(n0 + srow) * Dk + kc + scol0;
    u16* la = &At[srow * 72 + scol0];
    u16* lb = &Bs[srow * 72 + scol0];
#pragma unroll
    for (int c = 0; c < 4; ++c) {
      cp_g2l_b128(ga + c * 8, la + c * 8);
      cp_g2l_b128(gb + c * 8, lb + c * 8);
    }
    async_wait();
    __syncthreads();

#pragma unroll
    for (int ks = 0; ks < 2; ++ks) {
      FragAB af[2];
#pragma unroll
      for (int ms = 0; ms < 2; ++ms) {
        const int arow = wm + ms * 16 + lane16;
        const int ac0 = grp8 + ks * 32;      // A frag: chunks c0..c0+7, c0+16..c0+23
        af[ms].u[0] = *(const uint4*)&At[arow * 72 + ac0];
        af[ms].u[1] = *(const uint4*)&At[arow * 72 + ac0 + 16];
      }
#pragma unroll
      for (int ns = 0; ns < 4; ++ns) {
        FragAB bf_;
        const int brow = wn + ns * 16 + lane16;
        const int bc0 = grp8 + ks * 32;      // B frag: chunks c0..c0+7, c0+16..c0+23
        bf_.u[0] = *(const uint4*)&Bs[brow * 72 + bc0];
        bf_.u[1] = *(const uint4*)&Bs[brow * 72 + bc0 + 16];
#pragma unroll
        for (int ms = 0; ms < 2; ++ms)
          acc[ms][ns].v = __builtin_amdgcn_wmma_f32_16x16x32_bf16(
              false, af[ms].v, false, bf_.v, (short)0, acc[ms][ns].v, false, false);
      }
    }
    __syncthreads();
  }

  // ---- epilogue
#pragma unroll
  for (int ns = 0; ns < 4; ++ns) {
    const int n_global = n0 + wn + ns * 16 + lane16;
    const float bv = bias[n_global];
    if (mode == 0) {
      const int sel = n_global / D_MODEL;          // 0=q 1=k 2=v (uniform per block)
      const int d = n_global % D_MODEL;
      const int h = d >> 6, hh = d & (HDIM - 1);
#pragma unroll
      for (int ms = 0; ms < 2; ++ms)
#pragma unroll
        for (int r = 0; r < 8; ++r) {
          const int gm = m0 + wm + ms * 16 + r + grp8;
          const int b = gm >> 11, sidx = gm & (S_LEN - 1);
          const u16 hv = f32_to_bf16(acc[ms][ns].f[r] + bv);
          if (sel == 0)
            Qo[(((size_t)(b * NHEADS + h) * S_LEN) + sidx) * HDIM + hh] = hv;
          else if (sel == 1)
            Ko[(((size_t)(b * NHEADS + h) * S_LEN) + sidx) * HDIM + hh] = hv;
          else
            Vto[(((size_t)(b * NHEADS + h) * HDIM) + hh) * S_LEN + sidx] = hv;
        }
    } else {
#pragma unroll
      for (int ms = 0; ms < 2; ++ms)
#pragma unroll
        for (int r = 0; r < 8; ++r) {
          const int gm = m0 + wm + ms * 16 + r + grp8;
          out[(size_t)gm * D_MODEL + n_global] = acc[ms][ns].f[r] + bv;
        }
    }
  }
}

// ---------------------------------------------------------------- flash attention
// grid: x = q-tile (32), y = b*H+h (32). 128 threads = 4 waves; wave w owns
// q rows [q0+w*16, +16). K/V tiles staged once per block into LDS (async DMA);
// scores never touch HBM.
__global__ void flash_attn(const u16* __restrict__ Q, const u16* __restrict__ K,
                           const u16* __restrict__ Vt, u16* __restrict__ ctx) {
  __shared__ u16 Kt[64 * 72];          // key rows x hd
  __shared__ u16 Vs[64 * 72];          // hd rows x key
  __shared__ u16 Pp[4 * 16 * 72];      // per-wave P staging

  const int tid = threadIdx.x;
  const int wave = tid >> 5, lane = tid & 31;
  const int lane16 = lane & 15, grp = lane >> 4, grp8 = grp * 8;
  const int qt = blockIdx.x, bh = blockIdx.y;
  const int q0 = qt * 64;

  const u16* Qh = Q  + (size_t)bh * S_LEN * HDIM;
  const u16* Kh = K  + (size_t)bh * S_LEN * HDIM;
  const u16* Vh = Vt + (size_t)bh * HDIM * S_LEN;

  // Q A-fragments for this wave's 16 rows, K-dim = hd = 64 -> 2 fragments
  FragAB qa[2];
  const int qrow = q0 + wave * 16 + lane16;
#pragma unroll
  for (int ks = 0; ks < 2; ++ks) {
    const int c0 = grp8 + ks * 32;
    qa[ks].u[0] = *(const uint4*)(Qh + (size_t)qrow * HDIM + c0);
    qa[ks].u[1] = *(const uint4*)(Qh + (size_t)qrow * HDIM + c0 + 16);
  }

  float mrow[8], lrow[8];
  Acc8 o[4];
#pragma unroll
  for (int r = 0; r < 8; ++r) { mrow[r] = -1e30f; lrow[r] = 0.f; }
#pragma unroll
  for (int ns = 0; ns < 4; ++ns)
#pragma unroll
    for (int r = 0; r < 8; ++r) o[ns].f[r] = 0.f;

  const float scale = 0.125f;   // rsqrt(64)
  u16* pbase = &Pp[wave * 16 * 72];
  const int srow = tid >> 1, scol0 = (tid & 1) * 32;   // staging: 2 thr/row, 64B

  for (int j = 0; j <= qt; ++j) {   // causal: only key tiles <= query tile
    const int k0 = j * 64;

    // ---- stage K (64 keys x 64 hd) and V^T (64 hd x 64 keys) tiles
    {
      const u16* gk = Kh + (size_t)(k0 + srow) * HDIM + scol0;
      const u16* gv = Vh + (size_t)srow * S_LEN + k0 + scol0;
      u16* lk = &Kt[srow * 72 + scol0];
      u16* lv = &Vs[srow * 72 + scol0];
#pragma unroll
      for (int c = 0; c < 4; ++c) {
        cp_g2l_b128(gk + c * 8, lk + c * 8);
        cp_g2l_b128(gv + c * 8, lv + c * 8);
      }
      async_wait();
      __syncthreads();
    }

    // ---- S = Q @ K^T  (16 x 64 per wave)
    Acc8 s[4];
#pragma unroll
    for (int ns = 0; ns < 4; ++ns)
#pragma unroll
      for (int r = 0; r < 8; ++r) s[ns].f[r] = 0.f;

#pragma unroll
    for (int ks = 0; ks < 2; ++ks) {
      const int c0 = grp8 + ks * 32;
#pragma unroll
      for (int ns = 0; ns < 4; ++ns) {
        FragAB kb;
        const int krow = ns * 16 + lane16;
        kb.u[0] = *(const uint4*)&Kt[krow * 72 + c0];
        kb.u[1] = *(const uint4*)&Kt[krow * 72 + c0 + 16];
        s[ns].v = __builtin_amdgcn_wmma_f32_16x16x32_bf16(
            false, qa[ks].v, false, kb.v, (short)0, s[ns].v, false, false);
      }
    }

    // ---- scale + causal mask (diagonal tile only)
#pragma unroll
    for (int ns = 0; ns < 4; ++ns) {
      const int kg = k0 + ns * 16 + lane16;
#pragma unroll
      for (int r = 0; r < 8; ++r) {
        float v = s[ns].f[r] * scale;
        if (j == qt) {
          const int qg = q0 + wave * 16 + r + grp8;
          v = (kg <= qg) ? v : -1e10f;
        }
        s[ns].f[r] = v;
      }
    }

    // ---- online softmax: row max across 4 subtiles + 16 lanes (width-16 xor)
#pragma unroll
    for (int r = 0; r < 8; ++r) {
      float t = fmaxf(fmaxf(s[0].f[r], s[1].f[r]), fmaxf(s[2].f[r], s[3].f[r]));
      t = fmaxf(t, __shfl_xor(t, 8, 16));
      t = fmaxf(t, __shfl_xor(t, 4, 16));
      t = fmaxf(t, __shfl_xor(t, 2, 16));
      t = fmaxf(t, __shfl_xor(t, 1, 16));
      const float mnew = fmaxf(mrow[r], t);
      const float alpha = __expf(mrow[r] - mnew);
      mrow[r] = mnew;
      lrow[r] *= alpha;
#pragma unroll
      for (int ns = 0; ns < 4; ++ns) o[ns].f[r] *= alpha;
    }

    // ---- P = exp(S - m): row-sum partials + stage to LDS as bf16
    float psum[8];
#pragma unroll
    for (int r = 0; r < 8; ++r) psum[r] = 0.f;
#pragma unroll
    for (int ns = 0; ns < 4; ++ns) {
#pragma unroll
      for (int r = 0; r < 8; ++r) {
        const float p = __expf(s[ns].f[r] - mrow[r]);
        psum[r] += p;
        pbase[(r + grp8) * 72 + ns * 16 + lane16] = f32_to_bf16(p);
      }
    }
#pragma unroll
    for (int r = 0; r < 8; ++r) {
      float t = psum[r];
      t += __shfl_xor(t, 8, 16);
      t += __shfl_xor(t, 4, 16);
      t += __shfl_xor(t, 2, 16);
      t += __shfl_xor(t, 1, 16);
      lrow[r] += t;
    }

    // ---- O += P @ V   (contract over 64 keys -> 2 fragments)
#pragma unroll
    for (int ks = 0; ks < 2; ++ks) {
      FragAB pa;
      const int pc0 = grp8 + ks * 32;
      pa.u[0] = *(const uint4*)&pbase[lane16 * 72 + pc0];
      pa.u[1] = *(const uint4*)&pbase[lane16 * 72 + pc0 + 16];
      const int c0v = grp8 + ks * 32;
#pragma unroll
      for (int ns = 0; ns < 4; ++ns) {
        FragAB vb;
        const int vrow = ns * 16 + lane16;     // hd index
        vb.u[0] = *(const uint4*)&Vs[vrow * 72 + c0v];
        vb.u[1] = *(const uint4*)&Vs[vrow * 72 + c0v + 16];
        o[ns].v = __builtin_amdgcn_wmma_f32_16x16x32_bf16(
            false, pa.v, false, vb.v, (short)0, o[ns].v, false, false);
      }
    }
    __syncthreads();   // protect Kt/Vs before next tile's staging
  }

  // ---- normalize and store ctx [B*S, D] bf16 (A-layout for the final GEMM)
  const int b = bh / NHEADS, h = bh % NHEADS;
#pragma unroll
  for (int ns = 0; ns < 4; ++ns) {
    const int d = h * HDIM + ns * 16 + lane16;
#pragma unroll
    for (int r = 0; r < 8; ++r) {
      const int sidx = q0 + wave * 16 + r + grp8;
      const size_t grow = (size_t)(b * S_LEN + sidx);
      ctx[grow * D_MODEL + d] = f32_to_bf16(o[ns].f[r] / lrow[r]);
    }
  }
}

// ---------------------------------------------------------------- launch
extern "C" void kernel_launch(void* const* d_in, const int* in_sizes, int n_in,
                              void* d_out, int out_size, void* d_ws, size_t ws_size,
                              hipStream_t stream) {
  const float* x      = (const float*)d_in[0];
  const float* W_exp  = (const float*)d_in[1];
  const float* b_exp  = (const float*)d_in[2];
  const float* W_comp = (const float*)d_in[3];
  const float* b_comp = (const float*)d_in[4];
  float* out = (float*)d_out;

  // workspace carve (bf16 elements)
  const size_t nX   = (size_t)M_TOT * D_MODEL;               // 4M
  const size_t nWE  = (size_t)3 * D_MODEL * D_MODEL;         // 3M
  const size_t nWC  = (size_t)D_MODEL * D_MODEL;             // 1M
  const size_t nHK  = (size_t)BATCH * NHEADS * S_LEN * HDIM; // 4M
  u16* xb     = (u16*)d_ws;
  u16* WexpT  = xb + nX;
  u16* WcompT = WexpT + nWE;
  u16* Qb     = WcompT + nWC;
  u16* Kb     = Qb + nHK;
  u16* Vtb    = Kb + nHK;
  u16* ctxb   = Vtb + nHK;

  cvt_f32_bf16<<<(int)((nX + 255) / 256), 256, 0, stream>>>(x, xb, (int)nX);
  transpose_to_bf16<<<(int)(nWE / 256), 256, 0, stream>>>(W_exp, WexpT, 3 * D_MODEL, D_MODEL);
  transpose_to_bf16<<<(int)(nWC / 256), 256, 0, stream>>>(W_comp, WcompT, D_MODEL, D_MODEL);

  // QKV projection: M=4096, N=3072
  gemm_bf16<<<dim3(3 * D_MODEL / 128, M_TOT / 128), 256, 0, stream>>>(
      xb, WexpT, b_exp, 0, Qb, Kb, Vtb, nullptr);

  // flash attention: 32 q-tiles x 32 (b,h)
  flash_attn<<<dim3(S_LEN / 64, BATCH * NHEADS), 128, 0, stream>>>(Qb, Kb, Vtb, ctxb);

  // output projection: M=4096, N=1024, fp32 out
  gemm_bf16<<<dim3(D_MODEL / 128, M_TOT / 128), 256, 0, stream>>>(
      ctxb, WcompT, b_comp, 1, nullptr, nullptr, nullptr, out);
}
